// RWKV_46016279609999
// MI455X (gfx1250) — matrix operand order repaired and checked by reading, not compiled, not run on hardware
//
#include <hip/hip_runtime.h>
#include <hip/hip_bf16.h>

// ---------------- model constants ----------------
#define Bb   2
#define Tt   512
#define Ll   4
#define Cc   1024
#define Hh   16
#define Kk   64
#define Vv   32000
#define BT   (Bb*Tt)
#define D_DECAY 64
#define D_ICLR  32
#define D_V0    64
#define D_GATE  128

// ---------------- types ----------------
typedef __attribute__((ext_vector_type(16))) __bf16 v16bf;
typedef __attribute__((ext_vector_type(8)))  float  v8f;
typedef __attribute__((ext_vector_type(4)))  int    v4i;

__device__ __forceinline__ __bf16 f2bf(float f) {
  unsigned u = __builtin_bit_cast(unsigned, f);
  unsigned r = (u + 0x7FFFu + ((u >> 16) & 1u)) >> 16;   // round-to-nearest-even
  return __builtin_bit_cast(__bf16, (unsigned short)r);
}

// ---------------- CDNA5 async global->LDS copy (ASYNCcnt DMA) ----------------
#if defined(__HIP_DEVICE_COMPILE__)
# if __has_builtin(__builtin_amdgcn_global_load_async_to_lds_b128)
#  define HAS_ASYNC_LDS 1
# endif
# if __has_builtin(__builtin_amdgcn_s_wait_asynccnt)
#  define HAS_WAIT_ASYNC 1
# endif
#endif

#define GAS __attribute__((address_space(1)))
#define LAS __attribute__((address_space(3)))

__device__ __forceinline__ void copy16_g2l(const float* gp, float* lp) {
#ifdef HAS_ASYNC_LDS
  __builtin_amdgcn_global_load_async_to_lds_b128(
      (GAS v4i*)(gp), (LAS v4i*)(lp), 0, 0);
#else
  *(float4*)lp = *(const float4*)gp;
#endif
}
__device__ __forceinline__ void wait_async0() {
#ifdef HAS_WAIT_ASYNC
  __builtin_amdgcn_s_wait_asynccnt(0);
#elif defined(HAS_ASYNC_LDS)
  asm volatile("s_wait_asynccnt 0" ::: "memory");
#endif
}

// =====================================================================
// WMMA GEMM:  C[m,n] = sum_k A[m,k] * B'[n,k]
//   transB==0 : B stored row-major [N,K], ldb = row stride (NT form)
//   transB==1 : B stored row-major [K,N], ldb = row stride (NN form)
// Tile: 64(M) x 128(N) x 32(K), 8 waves; wave -> 16x64 strip (4 wmma,
// one shared A fragment). Pipeline: async DMA f32 tile -> LDS stage
// (double buffered) -> one-shot bf16 conversion -> v_wmma (f32 accum).
// =====================================================================
#define BM 64
#define BN 128
#define BK 32

__global__ __launch_bounds__(256)
void k_gemm(const float* __restrict__ A, int lda,
            const float* __restrict__ Bm, int ldb, int transB,
            float* __restrict__ Co, int ldc,
            int M, int N, int Kd)
{
  __shared__ float  sA[2][BM * BK];           // 2 x 8KB f32 stage
  __shared__ float  sB[2][BN * BK];           // 2 x 16KB f32 stage
  __shared__ __bf16 As16[BM][BK + 2];         // padded bf16 tiles
  __shared__ __bf16 Bs16[BN][BK + 2];

  const int tid  = threadIdx.x;
  const int lane = tid & 31;
  const int wave = tid >> 5;            // 0..7
  const int tileM = blockIdx.y * BM;
  const int tileN = blockIdx.x * BN;
  const int mi = (wave & 3) * 16;       // wave's M sub-tile
  const int ni = (wave >> 2) * 64;      // wave's N strip (4 x 16)
  const int hlf = lane >> 4;
  const int l16 = lane & 15;

  v8f acc[4];
  acc[0] = v8f{}; acc[1] = v8f{}; acc[2] = v8f{}; acc[3] = v8f{};

  const int nk = Kd / BK;               // all K dims are multiples of 32

  // ---- tile copy issue (async DMA when available) ----
  auto issue = [&](int ki, int buf) {
    const int k0 = ki * BK;
    // A tile: 64x32 f32 = 512 x 16B chunks, 2 per thread
    for (int c = tid; c < (BM * BK) / 4; c += 256) {
      const int r = c >> 3, kc = (c & 7) << 2;
      copy16_g2l(A + (long)(tileM + r) * lda + k0 + kc, &sA[buf][r * BK + kc]);
    }
    if (!transB) {
      // B tile: 128x32 f32 = 1024 x 16B chunks, 4 per thread
      for (int c = tid; c < (BN * BK) / 4; c += 256) {
        const int r = c >> 3, kc = (c & 7) << 2;
        float* lp = &sB[buf][r * BK + kc];
        const int n = tileN + r;
        if (n < N) copy16_g2l(Bm + (long)n * ldb + k0 + kc, lp);
        else { lp[0] = 0.f; lp[1] = 0.f; lp[2] = 0.f; lp[3] = 0.f; }
      }
    } else {
      // transposed weight [K,N]: strided gather, synchronous
      for (int e = tid; e < BN * BK; e += 256) {
        const int n = e & (BN - 1), k = e >> 7;
        const int gn = tileN + n;
        sB[buf][n * BK + k] = (gn < N) ? Bm[(long)(k0 + k) * ldb + gn] : 0.f;
      }
    }
  };

  issue(0, 0);
  for (int i = 0; i < nk; ++i) {
    const int cur = i & 1;
    wait_async0();
    __syncthreads();                    // stage[cur] visible to all waves
    if (i + 1 < nk) issue(i + 1, cur ^ 1);

    // ---- one-shot f32 -> bf16 conversion into padded tiles ----
    for (int e = tid; e < BM * BK; e += 256) {
      const int r = e >> 5, k = e & 31;
      As16[r][k] = f2bf(sA[cur][e]);
    }
    for (int e = tid; e < BN * BK; e += 256) {
      const int r = e >> 5, k = e & 31;
      Bs16[r][k] = f2bf(sB[cur][e]);
    }
    __syncthreads();                    // bf16 tiles ready

    // ---- fragments (contiguous k = 16*half + j) + 4 WMMA ----
    v16bf af, bf0, bf1, bf2, bf3;
#pragma unroll
    for (int j = 0; j < 16; ++j) {
      const int v  = j >> 1, p = j & 1;
      const int kA = (hlf << 4) | (v << 1) | p;
      const int kB = (hlf << 4) | (v << 1) | p;
      af[j]  = As16[mi + l16][kA];
      bf0[j] = Bs16[ni +  0 + l16][kB];
      bf1[j] = Bs16[ni + 16 + l16][kB];
      bf2[j] = Bs16[ni + 32 + l16][kB];
      bf3[j] = Bs16[ni + 48 + l16][kB];
    }
    acc[0] = __builtin_amdgcn_wmma_f32_16x16x32_bf16(false, af, false, bf0,
                                                     (short)0, acc[0], false, false);
    acc[1] = __builtin_amdgcn_wmma_f32_16x16x32_bf16(false, af, false, bf1,
                                                     (short)0, acc[1], false, false);
    acc[2] = __builtin_amdgcn_wmma_f32_16x16x32_bf16(false, af, false, bf2,
                                                     (short)0, acc[2], false, false);
    acc[3] = __builtin_amdgcn_wmma_f32_16x16x32_bf16(false, af, false, bf3,
                                                     (short)0, acc[3], false, false);
    __syncthreads();                    // done reading bf16 tiles
  }

  // ---- store C: VGPR r -> m = r + 8*half, n = lane&15 ----
#pragma unroll
  for (int s = 0; s < 4; ++s) {
    const int n0 = tileN + ni + s * 16 + l16;
    if (n0 < N) {
#pragma unroll
      for (int r = 0; r < 8; ++r) {
        const int m = tileM + mi + (hlf << 3) + r;
        if (m < M) Co[(long)m * ldc + n0] = acc[s][r];
      }
    }
  }
}

// =====================================================================
// LayerNorm (per row of C channels), 256 threads/row
// =====================================================================
__global__ __launch_bounds__(256)
void k_layernorm(const float* __restrict__ in, const float* __restrict__ w,
                 const float* __restrict__ b, float* __restrict__ out)
{
  const int row = blockIdx.x;
  const float* xr = in + (long)row * Cc;
  float s = 0.f, s2 = 0.f;
  for (int c = threadIdx.x; c < Cc; c += 256) { float v = xr[c]; s += v; s2 += v * v; }
  __shared__ float red0[256], red1[256];
  red0[threadIdx.x] = s; red1[threadIdx.x] = s2; __syncthreads();
  for (int st = 128; st > 0; st >>= 1) {
    if (threadIdx.x < st) { red0[threadIdx.x] += red0[threadIdx.x + st];
                            red1[threadIdx.x] += red1[threadIdx.x + st]; }
    __syncthreads();
  }
  const float mu = red0[0] * (1.f / Cc);
  const float var = red1[0] * (1.f / Cc) - mu * mu;
  const float rs = rsqrtf(var + 1e-5f);
  for (int c = threadIdx.x; c < Cc; c += 256)
    out[(long)row * Cc + c] = (xr[c] - mu) * rs * w[c] + b[c];
}

// Embedding gather + LayerNorm
__global__ __launch_bounds__(256)
void k_embed_ln(const int* __restrict__ tok, const float* __restrict__ emb,
                const float* __restrict__ w, const float* __restrict__ b,
                float* __restrict__ out)
{
  const int row = blockIdx.x;
  const float* xr = emb + (long)tok[row] * Cc;
  float s = 0.f, s2 = 0.f;
  for (int c = threadIdx.x; c < Cc; c += 256) { float v = xr[c]; s += v; s2 += v * v; }
  __shared__ float red0[256], red1[256];
  red0[threadIdx.x] = s; red1[threadIdx.x] = s2; __syncthreads();
  for (int st = 128; st > 0; st >>= 1) {
    if (threadIdx.x < st) { red0[threadIdx.x] += red0[threadIdx.x + st];
                            red1[threadIdx.x] += red1[threadIdx.x + st]; }
    __syncthreads();
  }
  const float mu = red0[0] * (1.f / Cc);
  const float var = red1[0] * (1.f / Cc) - mu * mu;
  const float rs = rsqrtf(var + 1e-5f);
  for (int c = threadIdx.x; c < Cc; c += 256)
    out[(long)row * Cc + c] = (xr[c] - mu) * rs * w[c] + b[c];
}

// token-shift lerp (in place over amt): amt <- xn + amt*(shift(xn)-xn)
__global__ void k_mix(const float* __restrict__ xn, float* __restrict__ amt,
                      int nmix)
{
  const long tot = (long)nmix * BT * Cc;
  const long i = (long)blockIdx.x * 256 + threadIdx.x;
  if (i >= tot) return;
  const int  c  = (int)(i % Cc);
  const long bt = (i / Cc) % BT;
  const int  t  = (int)(bt % Tt);
  const float xv = xn[bt * Cc + c];
  const float xp = (t > 0) ? xn[(bt - 1) * Cc + c] : 0.f;
  amt[i] = xv + amt[i] * (xp - xv);
}

__global__ void k_tanh_ip(float* __restrict__ x, long n) {
  const long i = (long)blockIdx.x * 256 + threadIdx.x;
  if (i < n) x[i] = tanhf(x[i]);
}
__global__ void k_sigmoid_ip(float* __restrict__ x, long n) {
  const long i = (long)blockIdx.x * 256 + threadIdx.x;
  if (i < n) x[i] = 1.f / (1.f + expf(-x[i]));
}
__global__ void k_relusq_ip(float* __restrict__ x, long n) {
  const long i = (long)blockIdx.x * 256 + threadIdx.x;
  if (i < n) { float v = fmaxf(x[i], 0.f); x[i] = v * v; }
}
__global__ void k_add(float* __restrict__ o, const float* __restrict__ a,
                      const float* __restrict__ b, long n) {
  const long i = (long)blockIdx.x * 256 + threadIdx.x;
  if (i < n) o[i] = a[i] + b[i];
}

// decay = exp(-exp(-0.5 - softplus(-(base + pre))))
__global__ void k_decayfin(float* __restrict__ dec, const float* __restrict__ pre,
                           const float* __restrict__ base, long n) {
  const long i = (long)blockIdx.x * 256 + threadIdx.x;
  if (i >= n) return;
  const int c = (int)(i % Cc);
  const float z  = base[c] + pre[i];
  const float zz = -z;
  const float sp = (zz > 20.f) ? zz : log1pf(expf(zz));  // softplus(-z)
  dec[i] = expf(-expf(-0.5f - sp));
}
// iclr = sigmoid(base + pre)
__global__ void k_iclrfin(float* __restrict__ ic, const float* __restrict__ pre,
                          const float* __restrict__ base, long n) {
  const long i = (long)blockIdx.x * 256 + threadIdx.x;
  if (i >= n) return;
  const int c = (int)(i % Cc);
  ic[i] = 1.f / (1.f + expf(-(base[c] + pre[i])));
}
// v += sigmoid(base + pre) * (v0 - v)
__global__ void k_v0fin(float* __restrict__ v, const float* __restrict__ v0,
                        const float* __restrict__ pre, const float* __restrict__ base,
                        long n) {
  const long i = (long)blockIdx.x * 256 + threadIdx.x;
  if (i >= n) return;
  const int c = (int)(i % Cc);
  const float m = 1.f / (1.f + expf(-(base[c] + pre[i])));
  v[i] = v[i] + m * (v0[i] - v[i]);
}
// k <- k + mix*(k*iclr - k)
__global__ void k_kup(float* __restrict__ k, const float* __restrict__ ic,
                      const float* __restrict__ mix, long n) {
  const long i = (long)blockIdx.x * 256 + threadIdx.x;
  if (i >= n) return;
  const int c = (int)(i % Cc);
  const float kv = k[i];
  k[i] = kv + mix[c] * (kv * ic[i] - kv);
}

// dk = normalize_per_head(k * dkey); one block per (bt,h), 64 threads
__global__ __launch_bounds__(64)
void k_dk(const float* __restrict__ kk, const float* __restrict__ dkey,
          float* __restrict__ dk)
{
  const int bt = blockIdx.x, h = blockIdx.y, j = threadIdx.x;
  const int c = h * Kk + j;
  const long idx = (long)bt * Cc + c;
  const float v = kk[idx] * dkey[c];
  __shared__ float red[64];
  red[j] = v * v; __syncthreads();
  for (int st = 32; st > 0; st >>= 1) {
    if (j < st) red[j] += red[j + st];
    __syncthreads();
  }
  const float nrm = fmaxf(sqrtf(red[0]), 1e-12f);
  dk[idx] = v / nrm;
}

// =====================================================================
// RWKV-7 sequential scan. One 64-thread block per (b,h); thread j owns
// state row v=j (64 f32 in VGPRs). Per-step k-vectors staged in LDS.
// =====================================================================
__global__ __launch_bounds__(64)
void k_scan(const float* __restrict__ r, const float* __restrict__ k,
            const float* __restrict__ v, const float* __restrict__ d,
            const float* __restrict__ ic, const float* __restrict__ dk,
            float* __restrict__ out)
{
  const int bh = blockIdx.x;
  const int b = bh / Hh, h = bh % Hh;
  const int j = threadIdx.x;
  float st[64];
#pragma unroll
  for (int q = 0; q < 64; ++q) st[q] = 0.f;
  __shared__ float ld[5][64];   // 0:d 1:dk 2:ic*dk 3:k 4:r

  for (int t = 0; t < Tt; ++t) {
    const long base = ((long)(b * Tt + t)) * Cc + h * Kk;
    ld[0][j] = d[base + j];
    ld[1][j] = dk[base + j];
    ld[2][j] = ic[base + j] * dk[base + j];
    ld[3][j] = k[base + j];
    ld[4][j] = r[base + j];
    const float vt = v[base + j];
    __syncthreads();

    float proj = 0.f;
#pragma unroll
    for (int q = 0; q < 64; ++q) { st[q] *= ld[0][q]; proj += st[q] * ld[1][q]; }
    float o = 0.f;
#pragma unroll
    for (int q = 0; q < 64; ++q) { st[q] += vt * ld[3][q] - proj * ld[2][q]; o += st[q] * ld[4][q]; }

    out[base + j] = o;
    __syncthreads();
  }
}

// GroupNorm(per head) + bonus + gate:  outm = (gn(out4) + (sum r*k*u)*v) * gate
__global__ __launch_bounds__(64)
void k_gnbonus(const float* __restrict__ o4, const float* __restrict__ r,
               const float* __restrict__ k, const float* __restrict__ v,
               const float* __restrict__ uvec, const float* __restrict__ gw,
               const float* __restrict__ gb, const float* __restrict__ gate,
               float* __restrict__ outm)
{
  const int bt = blockIdx.x, h = blockIdx.y, j = threadIdx.x;
  const int c = h * Kk + j;
  const long idx = (long)bt * Cc + c;
  const float g = o4[idx];
  const float rku = r[idx] * k[idx] * uvec[c];
  __shared__ float red0[64], red1[64], red2[64];
  red0[j] = g; red1[j] = g * g; red2[j] = rku; __syncthreads();
  for (int st = 32; st > 0; st >>= 1) {
    if (j < st) { red0[j] += red0[j + st]; red1[j] += red1[j + st]; red2[j] += red2[j + st]; }
    __syncthreads();
  }
  const float mu = red0[0] * (1.f / Kk);
  const float var = red1[0] * (1.f / Kk) - mu * mu;
  const float gn = (g - mu) * rsqrtf(var + 0.00064f) * gw[c] + gb[c];
  outm[idx] = (gn + red2[0] * v[idx]) * gate[idx];
}

// ---------------- input indices (setup_inputs dict order) ----------------
enum {
  IN_TOKENS = 0, IN_EMBED, IN_EMBED_LN_W, IN_EMBED_LN_B, IN_OUT_LN_W, IN_OUT_LN_B,
  IN_UNEMBED, IN_TM_LN_W, IN_TM_LN_B, IN_LERP_W, IN_DECAY_BASE, IN_DECAY_WA,
  IN_DECAY_WB, IN_ICLR_BASE, IN_ICLR_WA, IN_ICLR_WB, IN_V0_BASE, IN_V0_WA,
  IN_V0_WB, IN_WR, IN_WK, IN_WV, IN_WOUT, IN_DKEY, IN_GATE_W1, IN_GATE_W2,
  IN_ICLR_MIX, IN_U, IN_GN_W, IN_GN_B, IN_CM_LN_W, IN_CM_LN_B, IN_CM_LERP,
  IN_CM_WIN, IN_CM_WOUT
};

static inline dim3 eg(long n) { return dim3((unsigned)((n + 255) / 256)); }

static void gemm(const float* A, int lda, const float* B, int ldb, int transB,
                 float* C, int ldc, int M, int N, int K, hipStream_t s) {
  dim3 g((N + BN - 1) / BN, (M + BM - 1) / BM);
  k_gemm<<<g, 256, 0, s>>>(A, lda, B, ldb, transB, C, ldc, M, N, K);
}

extern "C" void kernel_launch(void* const* d_in, const int* in_sizes, int n_in,
                              void* d_out, int out_size, void* d_ws, size_t ws_size,
                              hipStream_t stream) {
  (void)in_sizes; (void)n_in; (void)out_size; (void)ws_size;
  const int*   tokens = (const int*)d_in[IN_TOKENS];
  const float* embed  = (const float*)d_in[IN_EMBED];

  const size_t BTC = (size_t)BT * Cc;
  float* W    = (float*)d_ws;
  float* x    = W;
  float* xn   = x    + BTC;
  float* v0   = xn   + BTC;
  float* mixd = v0   + BTC;        // 6*BTC
  float* rb   = mixd + 6 * BTC;
  float* kb   = rb   + BTC;
  float* vb   = kb   + BTC;
  float* dec  = vb   + BTC;
  float* icl  = dec  + BTC;
  float* dkb  = icl  + BTC;
  float* gat  = dkb  + BTC;
  float* o4   = gat  + BTC;
  float* mxo  = o4   + BTC;
  float* yb   = mxo  + BTC;
  float* tb   = yb   + BTC;        // BTC   (stage-2 temporaries)
  float* h4   = tb   + BTC;        // 4*BTC (channel mixer hidden)
  float* ts   = h4   + 4 * BTC;    // BT*128 (stage-1 temporaries)

  // embedding + input layernorm
  k_embed_ln<<<BT, 256, 0, stream>>>(tokens, embed,
      (const float*)d_in[IN_EMBED_LN_W], (const float*)d_in[IN_EMBED_LN_B], x);

  for (int l = 0; l < Ll; ++l) {
    const float* tm_ln_w = (const float*)d_in[IN_TM_LN_W] + (long)l * Cc;
    const float* tm_ln_b = (const float*)d_in[IN_TM_LN_B] + (long)l * Cc;
    const float* lerp_w  = (const float*)d_in[IN_LERP_W]  + (long)l * 6 * Cc * Cc;
    const float* Wr      = (const float*)d_in[IN_WR]      + (long)l * Cc * Cc;
    const float* Wk      = (const float*)d_in[IN_WK]      + (long)l * Cc * Cc;
    const float* Wv      = (const float*)d_in[IN_WV]      + (long)l * Cc * Cc;
    const float* Wout    = (const float*)d_in[IN_WOUT]    + (long)l * Cc * Cc;

    // xn = LN(x)
    k_layernorm<<<BT, 256, 0, stream>>>(x, tm_ln_w, tm_ln_b, xn);

    // amt_n = xn @ lerp_w[n].T ; then mixed = xn + amt*(shift - xn)
    for (int n = 0; n < 6; ++n)
      gemm(xn, Cc, lerp_w + (long)n * Cc * Cc, Cc, 0, mixd + (long)n * BTC, Cc,
           BT, Cc, Cc, stream);
    k_mix<<<eg((long)6 * BT * Cc), 256, 0, stream>>>(xn, mixd, 6);

    // r,k,v projections (mixed: 0=x_d 1=x_k 2=x_v 3=x_r 4=x_i 5=x_g)
    gemm(mixd + 3 * BTC, Cc, Wr, Cc, 0, rb, Cc, BT, Cc, Cc, stream);
    gemm(mixd + 1 * BTC, Cc, Wk, Cc, 0, kb, Cc, BT, Cc, Cc, stream);
    gemm(mixd + 2 * BTC, Cc, Wv, Cc, 0, vb, Cc, BT, Cc, Cc, stream);

    if (l == 0) {
      (void)hipMemcpyAsync(v0, vb, BTC * sizeof(float),
                           hipMemcpyDeviceToDevice, stream);
    } else {
      gemm(mixd + 2 * BTC, Cc, (const float*)d_in[IN_V0_WA] + (long)l * D_V0 * Cc,
           Cc, 0, ts, D_V0, BT, D_V0, Cc, stream);
      gemm(ts, D_V0, (const float*)d_in[IN_V0_WB] + (long)l * Cc * D_V0,
           D_V0, 0, tb, Cc, BT, Cc, D_V0, stream);
      k_v0fin<<<eg(BTC), 256, 0, stream>>>(vb, v0, tb,
          (const float*)d_in[IN_V0_BASE] + (long)l * Cc, (long)BTC);
    }

    // gate = sigmoid(x_g @ w1) @ w2   (both NN form -> transB=1)
    gemm(mixd + 5 * BTC, Cc, (const float*)d_in[IN_GATE_W1] + (long)l * Cc * D_GATE,
         D_GATE, 1, ts, D_GATE, BT, D_GATE, Cc, stream);
    k_sigmoid_ip<<<eg((long)BT * D_GATE), 256, 0, stream>>>(ts, (long)BT * D_GATE);
    gemm(ts, D_GATE, (const float*)d_in[IN_GATE_W2] + (long)l * D_GATE * Cc,
         Cc, 1, gat, Cc, BT, Cc, D_GATE, stream);

    // decay
    gemm(mixd + 0 * BTC, Cc, (const float*)d_in[IN_DECAY_WA] + (long)l * D_DECAY * Cc,
         Cc, 0, ts, D_DECAY, BT, D_DECAY, Cc, stream);
    k_tanh_ip<<<eg((long)BT * D_DECAY), 256, 0, stream>>>(ts, (long)BT * D_DECAY);
    gemm(ts, D_DECAY, (const float*)d_in[IN_DECAY_WB] + (long)l * Cc * D_DECAY,
         D_DECAY, 0, tb, Cc, BT, Cc, D_DECAY, stream);
    k_decayfin<<<eg(BTC), 256, 0, stream>>>(dec, tb,
        (const float*)d_in[IN_DECAY_BASE] + (long)l * Cc, (long)BTC);

    // dk from pre-update k
    k_dk<<<dim3(BT, Hh), 64, 0, stream>>>(kb,
        (const float*)d_in[IN_DKEY] + (long)l * Cc, dkb);

    // iclr
    gemm(mixd + 4 * BTC, Cc, (const float*)d_in[IN_ICLR_WA] + (long)l * D_ICLR * Cc,
         Cc, 0, ts, D_ICLR, BT, D_ICLR, Cc, stream);
    gemm(ts, D_ICLR, (const float*)d_in[IN_ICLR_WB] + (long)l * Cc * D_ICLR,
         D_ICLR, 0, tb, Cc, BT, Cc, D_ICLR, stream);
    k_iclrfin<<<eg(BTC), 256, 0, stream>>>(icl, tb,
        (const float*)d_in[IN_ICLR_BASE] + (long)l * Cc, (long)BTC);

    // k <- k + iclr_mix*(k*iclr - k)
    k_kup<<<eg(BTC), 256, 0, stream>>>(kb, icl,
        (const float*)d_in[IN_ICLR_MIX] + (long)l * Cc, (long)BTC);

    // sequential state scan
    k_scan<<<Bb * Hh, 64, 0, stream>>>(rb, kb, vb, dec, icl, dkb, o4);

    // group-norm + bonus + gate
    k_gnbonus<<<dim3(BT, Hh), 64, 0, stream>>>(o4, rb, kb, vb,
        (const float*)d_in[IN_U] + (long)l * Cc,
        (const float*)d_in[IN_GN_W] + (long)l * Cc,
        (const float*)d_in[IN_GN_B] + (long)l * Cc, gat, mxo);

    // out projection + residual (x = xn + mxo @ Wout.T)
    gemm(mxo, Cc, Wout, Cc, 0, yb, Cc, BT, Cc, Cc, stream);
    k_add<<<eg(BTC), 256, 0, stream>>>(x, xn, yb, (long)BTC);

    // ---------------- channel mixer ----------------
    k_layernorm<<<BT, 256, 0, stream>>>(x,
        (const float*)d_in[IN_CM_LN_W] + (long)l * Cc,
        (const float*)d_in[IN_CM_LN_B] + (long)l * Cc, xn);
    gemm(xn, Cc, (const float*)d_in[IN_CM_LERP] + (long)l * Cc * Cc,
         Cc, 0, tb, Cc, BT, Cc, Cc, stream);
    k_mix<<<eg(BTC), 256, 0, stream>>>(xn, tb, 1);
    gemm(tb, Cc, (const float*)d_in[IN_CM_WIN] + (long)l * 4 * Cc * Cc,
         Cc, 0, h4, 4 * Cc, BT, 4 * Cc, Cc, stream);
    k_relusq_ip<<<eg((long)4 * BTC), 256, 0, stream>>>(h4, (long)4 * BTC);
    gemm(h4, 4 * Cc, (const float*)d_in[IN_CM_WOUT] + (long)l * Cc * 4 * Cc,
         4 * Cc, 0, yb, Cc, BT, Cc, 4 * Cc, stream);
    k_add<<<eg(BTC), 256, 0, stream>>>(x, xn, yb, (long)BTC);
  }

  // final LN + unembed -> logits [BT, V]
  k_layernorm<<<BT, 256, 0, stream>>>(x, (const float*)d_in[IN_OUT_LN_W],
                                      (const float*)d_in[IN_OUT_LN_B], xn);
  gemm(xn, Cc, (const float*)d_in[IN_UNEMBED], Cc, 0, (float*)d_out, Vv,
       BT, Vv, Cc, stream);
}
